// NeuralFSDE_31945966748103
// MI455X (gfx1250) — matrix operand order repaired and checked by reading, not compiled, not run on hardware
//
#include <hip/hip_runtime.h>
#include <hip/hip_bf16.h>
#include <math.h>

typedef __bf16 bf16_t;
typedef __attribute__((ext_vector_type(16))) __bf16 v16bf;
typedef __attribute__((ext_vector_type(8)))  __bf16 v8bf;
typedef __attribute__((ext_vector_type(8)))  float  v8f;
typedef __attribute__((ext_vector_type(4)))  float  v4f;

// ---------------------------------------------------------------------------
// WMMA fragment loaders (wave32, V_WMMA_F32_16X16X32_BF16 layouts per ISA 7.12.2)
//
// A (16x32, 16-bit): lane row m = lane&15 for BOTH halves; half = lane>>4.
//   frag elems e=0..7  -> K = k0 + half*8 + e        (contiguous 8)
//   frag elems e=8..15 -> K = k0 + 16 + half*8 + (e-8) (contiguous 8)
// B (32x16, 16-bit): lane col n = lane&15; same K mapping as A (two chunks of 8)
//   => store weights transposed [N][K] and do two 16-byte contiguous loads
//      per fragment.
// C/D (16x16 f32): VGPR d, lane l -> M = d + half*8, N = lane&15.
// ---------------------------------------------------------------------------

__device__ __forceinline__ v16bf load_a_f32(const float* __restrict__ row, int k0, int half) {
  const float* p0 = row + k0 + half * 8;        // chunk of 8 f32 (32B, aligned)
  const float* p1 = p0 + 16;                    // second chunk of 8 f32
  v4f c00 = *(const v4f*)(p0);
  v4f c01 = *(const v4f*)(p0 + 4);
  v4f c10 = *(const v4f*)(p1);
  v4f c11 = *(const v4f*)(p1 + 4);
  v16bf a;
#pragma unroll
  for (int i = 0; i < 4; ++i) {
    a[i]      = (__bf16)c00[i];
    a[4 + i]  = (__bf16)c01[i];
    a[8 + i]  = (__bf16)c10[i];
    a[12 + i] = (__bf16)c11[i];
  }
  return a;
}

__device__ __forceinline__ v16bf load_a_bf16(const bf16_t* __restrict__ row, int k0, int half) {
  const bf16_t* p0 = row + k0 + half * 8;
  v8bf c0 = *(const v8bf*)(p0);        // 16B
  v8bf c1 = *(const v8bf*)(p0 + 16);   // 16B
  v16bf a;
#pragma unroll
  for (int i = 0; i < 8; ++i) { a[i] = c0[i]; a[8 + i] = c1[i]; }
  return a;
}

__device__ __forceinline__ v16bf load_b(const bf16_t* __restrict__ nrow, int k0, int half) {
  const bf16_t* p0 = nrow + k0 + half * 8;
  v8bf c0 = *(const v8bf*)(p0);        // 16B
  v8bf c1 = *(const v8bf*)(p0 + 16);   // 16B
  v16bf b;
#pragma unroll
  for (int i = 0; i < 8; ++i) { b[i] = c0[i]; b[8 + i] = c1[i]; }
  return b;
}

// ---------------------------------------------------------------------------
// One-time: convert weights f32 -> bf16, transposed to [N][K] row-major.
//   W1 (d,s): src [256,512] -> dst [512,256]
//   W2 (d,s): src [512,256] -> dst [256,512]
// 4 * 131072 elements, grid = 2048 x 256.
// ---------------------------------------------------------------------------
__global__ void fsde_prep(const float* __restrict__ Wd1, const float* __restrict__ Ws1,
                          const float* __restrict__ Wd2, const float* __restrict__ Ws2,
                          bf16_t* __restrict__ Wd1t, bf16_t* __restrict__ Ws1t,
                          bf16_t* __restrict__ Wd2t, bf16_t* __restrict__ Ws2t) {
  int idx   = blockIdx.x * 256 + threadIdx.x;   // 0 .. 524287
  int which = idx >> 17;
  int r     = idx & 131071;
  if (which < 2) {
    const float* src = which ? Ws1 : Wd1;       // [K=256][N=512]
    bf16_t*      dst = which ? Ws1t : Wd1t;     // [N=512][K=256]
    int n = r >> 8, k = r & 255;
    dst[r] = (bf16_t)src[k * 512 + n];
  } else {
    const float* src = (which == 3) ? Ws2 : Wd2; // [K=512][N=256]
    bf16_t*      dst = (which == 3) ? Ws2t : Wd2t; // [N=256][K=512]
    int n = r >> 9, k = r & 511;
    dst[r] = (bf16_t)src[k * 256 + n];
  }
}

// out[b, 0, s] = x0[b, s]; grid = 128 x 256 (exactly 32768 elems)
__global__ void fsde_init(const float* __restrict__ x0, float* __restrict__ out, int N) {
  int idx = blockIdx.x * 256 + threadIdx.x;
  int b = idx >> 8, s = idx & 255;
  out[(size_t)b * (size_t)(N + 1) * 256 + s] = x0[idx];
}

// ---------------------------------------------------------------------------
// Step kernel 1: h_{d,s} = tanh(x_t @ W1 + b1) as bf16, for both branches.
// x_t is read from the trajectory: out[b, t, k].
// 512 output tiles (2 branches * 8 Mtiles * 32 Ntiles), 1 wave per tile.
// grid = 64 blocks x 256 threads (8 waves/block).
// ---------------------------------------------------------------------------
__global__ __launch_bounds__(256)
void fsde_hidden(const float* __restrict__ out,
                 const bf16_t* __restrict__ Wd1t, const bf16_t* __restrict__ Ws1t,
                 const float* __restrict__ bd1, const float* __restrict__ bs1,
                 bf16_t* __restrict__ h_d, bf16_t* __restrict__ h_s,
                 int t, int N) {
  const int outLD = (N + 1) * 256;
  int wave = (blockIdx.x << 3) + (threadIdx.x >> 5);  // 0..511
  int lane = threadIdx.x & 31;
  int half = lane >> 4;
  int l15  = lane & 15;
  int branch = wave >> 8;                 // 0 = drift, 1 = diffusion
  int rem    = wave & 255;
  int tm = rem >> 5;                      // 0..7   (M tile over B=128)
  int tn = rem & 31;                      // 0..31  (N tile over Hd=512)

  const bf16_t* Wt   = branch ? Ws1t : Wd1t;
  const float*  b1   = branch ? bs1  : bd1;
  bf16_t*       hout = branch ? h_s  : h_d;

  int m = tm * 16 + l15;                  // batch row
  int n = tn * 16 + l15;                  // hidden col
  const float*  xrow = out + (size_t)m * outLD + (size_t)t * 256;  // x_t[m][*]
  const bf16_t* wrow = Wt + (size_t)n * 256;                       // W1t[n][*]

  v8f acc = {};
#pragma unroll
  for (int kb = 0; kb < 8; ++kb) {        // K = 256 = 8 * 32
    v16bf a = load_a_f32(xrow, kb * 32, half);
    v16bf b = load_b(wrow, kb * 32, half);
    acc = __builtin_amdgcn_wmma_f32_16x16x32_bf16(false, a, false, b,
                                                  (short)0, acc, false, false);
  }

  float bias = b1[n];
#pragma unroll
  for (int d = 0; d < 8; ++d) {
    int row = tm * 16 + d + half * 8;     // C/D layout: M = d + half*8
    hout[(size_t)row * 512 + n] = (bf16_t)tanhf(acc[d] + bias);
  }
}

// ---------------------------------------------------------------------------
// Step kernel 2: x_{t+1} = x_t + (h_d@Wd2 + bd2)*dt + (h_s@Ws2 + bs2)*noise,
// noise = fgn[t] * dt^H. Writes out[b, t+1, s].
// 128 output tiles (8 Mtiles * 16 Ntiles), 1 wave per tile; each wave does
// both branch GEMMs (2 * 16 WMMAs, K = 512).
// grid = 16 blocks x 256 threads.
// ---------------------------------------------------------------------------
__global__ __launch_bounds__(256)
void fsde_update(const bf16_t* __restrict__ h_d, const bf16_t* __restrict__ h_s,
                 const bf16_t* __restrict__ Wd2t, const bf16_t* __restrict__ Ws2t,
                 const float* __restrict__ bd2, const float* __restrict__ bs2,
                 const float* __restrict__ fgn, const float* __restrict__ raw_h,
                 float* __restrict__ out, int t, int N, float dt) {
  const int outLD = (N + 1) * 256;
  int wave = (blockIdx.x << 3) + (threadIdx.x >> 5);  // 0..127
  int lane = threadIdx.x & 31;
  int half = lane >> 4;
  int l15  = lane & 15;
  int tm = wave >> 4;                     // 0..7  (M tile over B=128)
  int tn = wave & 15;                     // 0..15 (N tile over S=256)

  int m = tm * 16 + l15;
  int n = tn * 16 + l15;
  const bf16_t* hdrow = h_d  + (size_t)m * 512;
  const bf16_t* hsrow = h_s  + (size_t)m * 512;
  const bf16_t* wdrow = Wd2t + (size_t)n * 512;
  const bf16_t* wsrow = Ws2t + (size_t)n * 512;

  v8f accD = {}, accS = {};
#pragma unroll
  for (int kb = 0; kb < 16; ++kb) {       // K = 512 = 16 * 32
    v16bf ad = load_a_bf16(hdrow, kb * 32, half);
    v16bf wd = load_b(wdrow, kb * 32, half);
    accD = __builtin_amdgcn_wmma_f32_16x16x32_bf16(false, ad, false, wd,
                                                   (short)0, accD, false, false);
    v16bf as = load_a_bf16(hsrow, kb * 32, half);
    v16bf ws = load_b(wsrow, kb * 32, half);
    accS = __builtin_amdgcn_wmma_f32_16x16x32_bf16(false, as, false, ws,
                                                   (short)0, accS, false, false);
  }

  // H = sigmoid(raw_h)*0.98 + 0.01 ; noise scale = dt^H
  float H     = 0.98f / (1.0f + expf(-raw_h[0])) + 0.01f;
  float scale = powf(dt, H);
  float biasD = bd2[n];
  float biasS = bs2[n];
  const float* fg = fgn + (size_t)t * 128 * 256;

#pragma unroll
  for (int d = 0; d < 8; ++d) {
    int row = tm * 16 + d + half * 8;     // batch index
    size_t base = (size_t)row * outLD + n;
    float xo    = out[base + (size_t)t * 256];
    float noise = fg[(size_t)row * 256 + n] * scale;
    float xn = xo + (accD[d] + biasD) * dt + (accS[d] + biasS) * noise;
    out[base + (size_t)(t + 1) * 256] = xn;
  }
}

// ---------------------------------------------------------------------------
// Host launcher. Shapes fixed by the reference: B=128, S=256, Hd=512.
// n_steps derived from in_sizes (host-readable) to stay capture-safe.
// ---------------------------------------------------------------------------
extern "C" void kernel_launch(void* const* d_in, const int* in_sizes, int n_in,
                              void* d_out, int out_size, void* d_ws, size_t ws_size,
                              hipStream_t stream) {
  const float* x0    = (const float*)d_in[0];
  const float* Wd1   = (const float*)d_in[1];
  const float* bd1   = (const float*)d_in[2];
  const float* Wd2   = (const float*)d_in[3];
  const float* bd2   = (const float*)d_in[4];
  const float* Ws1   = (const float*)d_in[5];
  const float* bs1   = (const float*)d_in[6];
  const float* Ws2   = (const float*)d_in[7];
  const float* bs2   = (const float*)d_in[8];
  const float* raw_h = (const float*)d_in[9];
  const float* fgn   = (const float*)d_in[10];

  const int B = 128, S = 256;
  const int N = in_sizes[10] / (B * S);     // fgn is [N, B, S]
  const float dt = 1.0f / (float)N;

  float* out = (float*)d_out;
  char*  ws  = (char*)d_ws;
  bf16_t* Wd1t = (bf16_t*)(ws + 0);         // [512][256] bf16  (256 KB)
  bf16_t* Ws1t = (bf16_t*)(ws + (1u << 18));
  bf16_t* Wd2t = (bf16_t*)(ws + (2u << 18)); // [256][512] bf16
  bf16_t* Ws2t = (bf16_t*)(ws + (3u << 18));
  bf16_t* h_d  = (bf16_t*)(ws + (4u << 18)); // [128][512] bf16 (128 KB)
  bf16_t* h_s  = (bf16_t*)(ws + (4u << 18) + (1u << 17));

  fsde_prep<<<2048, 256, 0, stream>>>(Wd1, Ws1, Wd2, Ws2, Wd1t, Ws1t, Wd2t, Ws2t);
  fsde_init<<<128, 256, 0, stream>>>(x0, out, N);

  for (int t = 0; t < N; ++t) {
    fsde_hidden<<<64, 256, 0, stream>>>(out, Wd1t, Ws1t, bd1, bs1, h_d, h_s, t, N);
    fsde_update<<<16, 256, 0, stream>>>(h_d, h_s, Wd2t, Ws2t, bd2, bs2,
                                        fgn, raw_h, out, t, N, dt);
  }
}
